// GraphConvolution_74071005986924
// MI455X (gfx1250) — matrix operand order repaired and checked by reading, not compiled, not run on hardware
//
#include <hip/hip_runtime.h>

// ---- CDNA5 WMMA types ----
typedef __attribute__((ext_vector_type(16))) __bf16 v16bf;
typedef __attribute__((ext_vector_type(8)))  float  v8f;
typedef __attribute__((ext_vector_type(4)))  unsigned int u32x4;

#define D_DIM 512      // D_IN == D_OUT == 512
#define N2    1024     // fused [w0|w1] column count

static __device__ __forceinline__ unsigned short f2bf(float f) {
    // round-to-nearest-even f32 -> bf16
    unsigned u = __float_as_uint(f);
    unsigned r = u + 0x7FFFu + ((u >> 16) & 1u);
    return (unsigned short)(r >> 16);
}

// ---------- x (f32 [N,512]) -> xb (bf16 [Mpad,512]), zero-pad tail rows ----------
__global__ void cvt_x_kernel(const float* __restrict__ x, unsigned short* __restrict__ xb,
                             int n_nodes, int mpad) {
    size_t i = (size_t)blockIdx.x * blockDim.x + threadIdx.x;   // 8 elems per thread
    size_t total = (size_t)mpad * (D_DIM / 8);
    if (i >= total) return;
    size_t base = i * 8;
    int row = (int)(base >> 9);
    union { unsigned short s[8]; u32x4 q; } o;
    if (row < n_nodes) {
        const float* p = x + base;
#pragma unroll
        for (int j = 0; j < 8; ++j) o.s[j] = f2bf(p[j]);
    } else {
#pragma unroll
        for (int j = 0; j < 8; ++j) o.s[j] = 0;
    }
    *(u32x4*)(xb + base) = o.q;
}

// ---------- [w0|w1] f32 -> wb bf16 [512,1024] row-major (K-major, N contiguous) ----------
__global__ void cvt_w_kernel(const float* __restrict__ w0, const float* __restrict__ w1,
                             unsigned short* __restrict__ wb) {
    int i = blockIdx.x * blockDim.x + threadIdx.x;      // 65536 threads, 8 elems each
    if (i >= D_DIM * N2 / 8) return;
    int k  = i >> 7;            // 1024/8 = 128 chunks per k-row
    int n0 = (i & 127) * 8;
    const float* src = (n0 < D_DIM) ? (w0 + (size_t)k * D_DIM + n0)
                                    : (w1 + (size_t)k * D_DIM + (n0 - D_DIM));
    union { unsigned short s[8]; u32x4 q; } o;
#pragma unroll
    for (int j = 0; j < 8; ++j) o.s[j] = f2bf(src[j]);
    *(u32x4*)(wb + (size_t)k * N2 + n0) = o.q;
}

// ---------- pre[Mpad,1024] = xb @ wb  (bf16 WMMA, f32 accumulate) ----------
// Block: 128 rows x 128 cols; 8 waves = 4(M) x 2(N); wave: 32x64 via 2x4 accumulators.
__global__ __launch_bounds__(256) void gemm_wmma_kernel(
        const unsigned short* __restrict__ xb,   // bf16 bits [Mpad,512]
        const unsigned short* __restrict__ wb,   // bf16 bits [512,1024]
        float* __restrict__ pre) {
    const int lane = threadIdx.x & 31;
    const int wid  = threadIdx.x >> 5;
    const int mw   = wid >> 1;                 // 0..3
    const int nw   = wid & 1;                  // 0..1
    const int m0   = blockIdx.x * 128 + mw * 32;
    const int n0   = blockIdx.y * 128 + nw * 64;

    // A-fragment addressing (ISA 16-bit A 16x32 layout):
    //   lane L holds row (L&15); elems 0..7  = K kk+8*(L>>4)+0..7
    //                            elems 8..15 = K kk+16+8*(L>>4)+0..7
    const unsigned short* xrow0 = xb + (size_t)(m0 + (lane & 15)) * D_DIM;
    const unsigned short* xrow1 = xrow0 + (size_t)16 * D_DIM;
    const int klo = (lane >> 4) * 8;

    union Frag { v16bf v; u32x4 q[2]; };
    v8f acc0[4] = {};
    v8f acc1[4] = {};

    for (int kk = 0; kk < D_DIM; kk += 32) {
        // Load ALL fragments for this K-step into distinct registers first, so
        // the compiler can issue one load clause and stagger s_wait_loadcnt.
        Frag A0, A1, B[4];
        A0.q[0] = *(const u32x4*)(xrow0 + kk + klo);
        A0.q[1] = *(const u32x4*)(xrow0 + kk + 16 + klo);
        A1.q[0] = *(const u32x4*)(xrow1 + kk + klo);
        A1.q[1] = *(const u32x4*)(xrow1 + kk + 16 + klo);
        // B-fragment: lane L holds col n0+j*16+(L&15); elems 0..7 = K kk+8*(L>>4)+0..7, elems 8..15 = K kk+16+8*(L>>4)+0..7
        const unsigned short* wr = wb + (size_t)kk * N2 + n0 + (lane & 15);
#pragma unroll
        for (int j = 0; j < 4; ++j) {
#pragma unroll
            for (int i = 0; i < 4; ++i) {
                const int k0 = klo + 2 * i;
                B[j].q[0][i] = (unsigned)wr[(size_t)k0 * N2 + j * 16] | ((unsigned)wr[(size_t)(k0 + 1) * N2 + j * 16] << 16);
                B[j].q[1][i] = (unsigned)wr[(size_t)(16 + k0) * N2 + j * 16] | ((unsigned)wr[(size_t)(17 + k0) * N2 + j * 16] << 16);
            }
        }
#pragma unroll
        for (int j = 0; j < 4; ++j) {
            acc0[j] = __builtin_amdgcn_wmma_f32_16x16x32_bf16(
                false, A0.v, false, B[j].v, (short)0, acc0[j], false, false);
            acc1[j] = __builtin_amdgcn_wmma_f32_16x16x32_bf16(
                false, A1.v, false, B[j].v, (short)0, acc1[j], false, false);
        }
    }

    // C/D layout: lane L -> col n0+j*16+(L&15); VGPR r -> row m0 + r + 8*(L>>4)
    float* prow0 = pre + (size_t)(m0 + (lane >> 4) * 8) * N2 + n0 + (lane & 15);
    float* prow1 = prow0 + (size_t)16 * N2;
#pragma unroll
    for (int j = 0; j < 4; ++j)
#pragma unroll
        for (int r = 0; r < 8; ++r) {
            prow0[(size_t)r * N2 + j * 16] = acc0[j][r];
            prow1[(size_t)r * N2 + j * 16] = acc1[j][r];
        }
}

// ---------- zero d_out ----------
__global__ void zero_kernel(float* __restrict__ p, size_t n4) {
    size_t i = (size_t)blockIdx.x * blockDim.x + threadIdx.x;
    if (i >= n4) return;
    u32x4 z = {0u, 0u, 0u, 0u};
    *(u32x4*)(p + i * 4) = z;
}

// ---------- SpMM scatter: one block per (edge, support) ----------
__global__ __launch_bounds__(256) void spmm_kernel(
        const float* __restrict__ pre,
        const int* __restrict__ r0, const int* __restrict__ c0, const float* __restrict__ v0,
        const int* __restrict__ r1, const int* __restrict__ c1, const float* __restrict__ v1,
        float* __restrict__ out, int n_edges) {
    int e = blockIdx.x;
    if (e >= n_edges) return;
    int s = blockIdx.y;
    int row, col; float val; int doff;
    if (s == 0) { row = r0[e]; col = c0[e]; val = v0[e]; doff = 0; }
    else        { row = r1[e]; col = c1[e]; val = v1[e]; doff = D_DIM; }

    const float* src = pre + (size_t)col * N2 + doff;
    float* dst = out + (size_t)row * D_DIM;
    int t = threadIdx.x;
    float a0 = src[t]       * val;
    float a1 = src[t + 256] * val;
    __hip_atomic_fetch_add(&dst[t],       a0, __ATOMIC_RELAXED, __HIP_MEMORY_SCOPE_AGENT);
    __hip_atomic_fetch_add(&dst[t + 256], a1, __ATOMIC_RELAXED, __HIP_MEMORY_SCOPE_AGENT);
}

// ---------- out = relu(out + b) ----------
__global__ void bias_relu_kernel(float* __restrict__ out, const float* __restrict__ b, size_t n4) {
    size_t i = (size_t)blockIdx.x * blockDim.x + threadIdx.x;
    if (i >= n4) return;
    float* p = out + i * 4;
    int col = (int)((i * 4) & (D_DIM - 1));
#pragma unroll
    for (int j = 0; j < 4; ++j) {
        float v = p[j] + b[col + j];
        p[j] = v > 0.0f ? v : 0.0f;
    }
}

extern "C" void kernel_launch(void* const* d_in, const int* in_sizes, int n_in,
                              void* d_out, int out_size, void* d_ws, size_t ws_size,
                              hipStream_t stream) {
    const float* x     = (const float*)d_in[0];
    const float* w0    = (const float*)d_in[1];
    const float* w1    = (const float*)d_in[2];
    const float* b     = (const float*)d_in[3];
    const float* vals0 = (const float*)d_in[4];
    const float* vals1 = (const float*)d_in[5];
    const int*   rows0 = (const int*)d_in[6];
    const int*   cols0 = (const int*)d_in[7];
    const int*   rows1 = (const int*)d_in[8];
    const int*   cols1 = (const int*)d_in[9];

    const int n_nodes = in_sizes[0] / D_DIM;
    const int n_edges = in_sizes[4];
    const int mpad    = (n_nodes + 127) & ~127;   // M tile = 128

    // Workspace layout (~246 MB):
    unsigned short* xb = (unsigned short*)d_ws;                 // bf16 [mpad,512]
    unsigned short* wb = xb + (size_t)mpad * D_DIM;             // bf16 [512,1024]
    float*          pre = (float*)(wb + (size_t)D_DIM * N2);    // f32  [mpad,1024]

    // 1) convert inputs to bf16
    {
        size_t total = (size_t)mpad * (D_DIM / 8);
        cvt_x_kernel<<<dim3((unsigned)((total + 255) / 256)), dim3(256), 0, stream>>>(
            x, xb, n_nodes, mpad);
        cvt_w_kernel<<<dim3((D_DIM * N2 / 8 + 255) / 256), dim3(256), 0, stream>>>(w0, w1, wb);
    }

    // 2) zero output accumulator
    size_t n4 = (size_t)n_nodes * D_DIM / 4;
    zero_kernel<<<dim3((unsigned)((n4 + 255) / 256)), dim3(256), 0, stream>>>((float*)d_out, n4);

    // 3) fused GEMM: pre = xb @ [w0|w1]
    gemm_wmma_kernel<<<dim3(mpad / 128, N2 / 128), dim3(256), 0, stream>>>(xb, wb, pre);

    // 4) scatter-add both supports
    spmm_kernel<<<dim3((unsigned)n_edges, 2), dim3(256), 0, stream>>>(
        pre, rows0, cols0, vals0, rows1, cols1, vals1, (float*)d_out, n_edges);

    // 5) bias + relu
    bias_relu_kernel<<<dim3((unsigned)((n4 + 255) / 256)), dim3(256), 0, stream>>>(
        (float*)d_out, b, n4);
}
